// MoELayer_SMEAR_25202868093319
// MI455X (gfx1250) — matrix operand order repaired and checked by reading, not compiled, not run on hardware
//
#include <hip/hip_runtime.h>
#include <hip/hip_bf16.h>
#include <stdint.h>

// ---------------------------------------------------------------------------
// MoE-SMEAR on MI455X (gfx1250, wave32).
// Dominant cost: two batched bf16 GEMMs (2048x1024x4096, 2048x4096x1024, x8)
// -> ~275 GFLOP, compute-bound on the v_wmma_f32_16x16x32_bf16 path.
// GEMM: block tile 128x256x32, 8 waves, 64x64 wave tile (16 WMMA / k-step),
// double-buffered LDS filled by GLOBAL_LOAD_ASYNC_TO_LDS_B128 (ASYNCcnt).
// ---------------------------------------------------------------------------

typedef __bf16 bf16;
typedef __attribute__((ext_vector_type(16))) __bf16 v16bf;
typedef __attribute__((ext_vector_type(8)))  __bf16 v8bf;
typedef __attribute__((ext_vector_type(4)))  __bf16 v4bf;
typedef __attribute__((ext_vector_type(8)))  float  v8f;
typedef int v4i __attribute__((vector_size(16)));

constexpr int Bd = 8;      // batch
constexpr int Ld = 2048;   // tokens
constexpr int Dd = 1024;   // model dim
constexpr int Hd = 4096;   // hidden dim
constexpr int Ed = 8;      // experts

// ---- gfx1250 async-to-LDS plumbing ----------------------------------------
#if __has_builtin(__builtin_amdgcn_global_load_async_to_lds_b128)
#define HAVE_ASYNC_LDS 1
typedef __attribute__((address_space(1))) v4i gv4i;   // global int4
typedef __attribute__((address_space(3))) v4i lv4i;   // LDS int4
__device__ __forceinline__ gv4i* gbl_cast(const void* p) { return (gv4i*)p; }
__device__ __forceinline__ lv4i* lds_cast(void* p)       { return (lv4i*)p; }
#define ASYNC_B128(g, l, off) \
    __builtin_amdgcn_global_load_async_to_lds_b128(gbl_cast(g), lds_cast(l), off, 0)
#if __has_builtin(__builtin_amdgcn_s_wait_asynccnt)
#define WAIT_ASYNC(n) __builtin_amdgcn_s_wait_asynccnt(n)
#else
#define WAIT_ASYNC(n) asm volatile("s_wait_asynccnt %0" ::"n"(n) : "memory")
#endif
#else
#define HAVE_ASYNC_LDS 0
#endif

// ---------------------------------------------------------------------------
// Router: per (b, 256-token chunk): logits -> softmax -> mask -> partial sums.
// Deterministic (tree reduction to per-chunk partials, no float atomics).
// ---------------------------------------------------------------------------
__global__ __launch_bounds__(256) void router_kernel(
    const float* __restrict__ x, const float* __restrict__ mask,
    const float* __restrict__ rw, const float* __restrict__ rb,
    float* __restrict__ part_up /*[B][8][E]*/, float* __restrict__ part_ms /*[B][8]*/)
{
    __shared__ float smem[Ed * Dd];          // 32 KB: router weights, reused for reduction
    const int tid = threadIdx.x;
    const int b = blockIdx.x, chunk = blockIdx.y;

    const float4* rw4 = (const float4*)rw;
    float4* sm4 = (float4*)smem;
#pragma unroll
    for (int i = 0; i < (Ed * Dd / 4) / 256; ++i) sm4[tid + i * 256] = rw4[tid + i * 256];
    __syncthreads();

    const int l = chunk * 256 + tid;
    const float4* xr = (const float4*)(x + ((size_t)b * Ld + l) * Dd);

    float acc[Ed];
#pragma unroll
    for (int e = 0; e < Ed; ++e) acc[e] = 0.f;

    for (int d4 = 0; d4 < Dd / 4; ++d4) {
        float4 xv = xr[d4];
#pragma unroll
        for (int e = 0; e < Ed; ++e) {
            const float* w = &smem[e * Dd + d4 * 4];
            acc[e] += xv.x * w[0] + xv.y * w[1] + xv.z * w[2] + xv.w * w[3];
        }
    }

    float mx = -3.4e38f;
#pragma unroll
    for (int e = 0; e < Ed; ++e) { acc[e] += rb[e]; mx = fmaxf(mx, acc[e]); }
    float ssum = 0.f;
#pragma unroll
    for (int e = 0; e < Ed; ++e) { acc[e] = __expf(acc[e] - mx); ssum += acc[e]; }
    const float mval = mask[(size_t)b * Ld + l];
    const float inv = mval / ssum;
#pragma unroll
    for (int e = 0; e < Ed; ++e) acc[e] *= inv;

    __syncthreads();
    float* red = smem;
#pragma unroll
    for (int e = 0; e < Ed; ++e) {
        red[tid] = acc[e];
        __syncthreads();
        for (int st = 128; st > 0; st >>= 1) {
            if (tid < st) red[tid] += red[tid + st];
            __syncthreads();
        }
        if (tid == 0) part_up[((size_t)b * 8 + chunk) * Ed + e] = red[0];
        __syncthreads();
    }
    red[tid] = mval;
    __syncthreads();
    for (int st = 128; st > 0; st >>= 1) {
        if (tid < st) red[tid] += red[tid + st];
        __syncthreads();
    }
    if (tid == 0) part_ms[b * 8 + chunk] = red[0];
}

__global__ void finalize_up_kernel(const float* __restrict__ part_up,
                                   const float* __restrict__ part_ms,
                                   float* __restrict__ up /*[B][E]*/)
{
    const int t = threadIdx.x;
    if (t >= Bd * Ed) return;
    const int b = t >> 3, e = t & 7;
    float s = 0.f, m = 0.f;
#pragma unroll
    for (int c = 0; c < 8; ++c) {
        s += part_up[((size_t)b * 8 + c) * Ed + e];
        m += part_ms[b * 8 + c];
    }
    up[t] = s / fmaxf(m, 1.0f);
}

// mb[b][n] = sum_e up[b][e] * bsrc[e][n]
__global__ __launch_bounds__(256) void merge_bias_kernel(
    const float* __restrict__ up, const float* __restrict__ bsrc,
    float* __restrict__ dst, int N)
{
    const int i = blockIdx.x * 256 + threadIdx.x;
    const int b = i / N, n = i - b * N;
    float s = 0.f;
#pragma unroll
    for (int e = 0; e < Ed; ++e) s += up[b * Ed + e] * bsrc[(size_t)e * N + n];
    dst[i] = s;
}

// dst[b][i] = bf16( sum_e up[b][e] * Wsrc[e][i] ), vectorized x4.
__global__ __launch_bounds__(256) void merge_w_kernel(
    const float* __restrict__ up, const float* __restrict__ Wsrc,
    bf16* __restrict__ dst, int S)
{
    const size_t i4 = ((size_t)blockIdx.x * 256 + threadIdx.x) * 4;
    float4 w[Ed];
#pragma unroll
    for (int e = 0; e < Ed; ++e) w[e] = *(const float4*)(Wsrc + (size_t)e * S + i4);
#pragma unroll
    for (int b = 0; b < Bd; ++b) {
        float ox = 0.f, oy = 0.f, oz = 0.f, ow = 0.f;
#pragma unroll
        for (int e = 0; e < Ed; ++e) {
            const float u = up[b * Ed + e];
            ox += u * w[e].x; oy += u * w[e].y; oz += u * w[e].z; ow += u * w[e].w;
        }
        v4bf o;
        o[0] = (bf16)ox; o[1] = (bf16)oy; o[2] = (bf16)oz; o[3] = (bf16)ow;
        *(v4bf*)(dst + (size_t)b * S + i4) = o;
    }
}

__global__ __launch_bounds__(256) void convert_f32_bf16_kernel(
    const float* __restrict__ in, bf16* __restrict__ outb)
{
    const size_t i4 = ((size_t)blockIdx.x * 256 + threadIdx.x) * 4;
    float4 v = *(const float4*)(in + i4);
    v4bf o;
    o[0] = (bf16)v.x; o[1] = (bf16)v.y; o[2] = (bf16)v.z; o[3] = (bf16)v.w;
    *(v4bf*)(outb + i4) = o;
}

// ---------------------------------------------------------------------------
// Tiled WMMA GEMM:  C[M,N] = A[M,K] * W[N,K]^T + bias[N]  (optional ReLU)
// Block tile 128x256x32; 8 waves in 2x4 grid, wave tile 64x64 (4x4 WMMA).
// LDS rows padded to 40 elems (80 B = 20 banks) -> conflict-free 16 B reads.
// Double-buffered LDS, filled via async-to-LDS when available.
// ---------------------------------------------------------------------------
constexpr int BM = 128, BN = 256, BK = 32;
constexpr int LSTR = BK + 8;   // 40 bf16 per LDS row

template <bool RELU, bool OUTBF>
__global__ __launch_bounds__(256) void gemm_bf16_kernel(
    const bf16* __restrict__ A, const bf16* __restrict__ W,
    const float* __restrict__ bias, void* __restrict__ outp,
    int M, int N, int K)
{
    __shared__ alignas(16) bf16 As[2][BM * LSTR];   // 2 x 10 KB
    __shared__ alignas(16) bf16 Ws[2][BN * LSTR];   // 2 x 20 KB

    const int tid = threadIdx.x;
    const int b = blockIdx.z;
    A    += (size_t)b * M * K;
    W    += (size_t)b * N * K;
    bias += (size_t)b * N;

    const int bn0 = blockIdx.x * BN;
    const int bm0 = blockIdx.y * BM;

    // Staging map. A tile: thread -> (row = tid/2, 32B half of the 64B row).
    //              W tile: thread -> full 64B row tid.
    const int rowA = tid >> 1;
    const int colA = (tid & 1) * 16;                   // element offset (2 x 16B segs)
    const bf16* gA = A + (size_t)(bm0 + rowA) * K + colA;
    const bf16* gW = W + (size_t)(bn0 + tid) * K;
    const int laOff = rowA * LSTR + colA;
    const int lwOff = tid * LSTR;

    const int wave = tid >> 5;
    const int lane = tid & 31;
    const int lh = lane & 15, hh = lane >> 4;
    const int wm0 = (wave >> 2) * 64;   // 2 wave rows (M)
    const int wn0 = (wave & 3) * 64;    // 4 wave cols (N)

    v8f acc[4][4] = {};
    const int nkt = K / BK;

#if HAVE_ASYNC_LDS
    // Per stage: 2 async b128 (A) + 4 async b128 (W) per thread, in-order.
    {
        const bf16 *pa = gA, *pw = gW;
        ASYNC_B128(pa, &As[0][laOff], 0);  ASYNC_B128(pa, &As[0][laOff], 16);
        ASYNC_B128(pw, &Ws[0][lwOff], 0);  ASYNC_B128(pw, &Ws[0][lwOff], 16);
        ASYNC_B128(pw, &Ws[0][lwOff], 32); ASYNC_B128(pw, &Ws[0][lwOff], 48);
    }
    for (int kt = 0; kt < nkt; ++kt) {
        const int cur = kt & 1;
        if (kt + 1 < nkt) {                   // prefetch next stage into other buffer
            const bf16* pa = gA + (size_t)(kt + 1) * BK;
            const bf16* pw = gW + (size_t)(kt + 1) * BK;
            ASYNC_B128(pa, &As[cur ^ 1][laOff], 0);  ASYNC_B128(pa, &As[cur ^ 1][laOff], 16);
            ASYNC_B128(pw, &Ws[cur ^ 1][lwOff], 0);  ASYNC_B128(pw, &Ws[cur ^ 1][lwOff], 16);
            ASYNC_B128(pw, &Ws[cur ^ 1][lwOff], 32); ASYNC_B128(pw, &Ws[cur ^ 1][lwOff], 48);
            WAIT_ASYNC(6);                    // stage kt's 6 loads retired (in-order)
        } else {
            WAIT_ASYNC(0);
        }
        __syncthreads();                      // all waves' stage-kt fills visible

        v16bf aF[4], wF[4];
#pragma unroll
        for (int mi = 0; mi < 4; ++mi) {
            const bf16* p = &As[cur][(wm0 + mi * 16 + lh) * LSTR + hh * 8];
            v8bf lo = *(const v8bf*)p;
            v8bf hi = *(const v8bf*)(p + 16);
            aF[mi] = __builtin_shufflevector(lo, hi, 0,1,2,3,4,5,6,7,8,9,10,11,12,13,14,15);
        }
#pragma unroll
        for (int ni = 0; ni < 4; ++ni) {
            const bf16* p = &Ws[cur][(wn0 + ni * 16 + lh) * LSTR + hh * 8];
            v8bf lo = *(const v8bf*)p;
            v8bf hi = *(const v8bf*)(p + 16);
            wF[ni] = __builtin_shufflevector(lo, hi, 0,1,2,3,4,5,6,7,8,9,10,11,12,13,14,15);
        }
#pragma unroll
        for (int mi = 0; mi < 4; ++mi)
#pragma unroll
            for (int ni = 0; ni < 4; ++ni)
                acc[mi][ni] = __builtin_amdgcn_wmma_f32_16x16x32_bf16(
                    false, aF[mi], false, wF[ni], (short)0, acc[mi][ni], false, false);
        __syncthreads();                      // done reading buf before its re-fill
    }
#else
    // Fallback: register staging -> ds_store, as in the classic pipeline.
    uint4 aR0 = *(const uint4*)gA;
    uint4 aR1 = *(const uint4*)(gA + 8);
    uint4 wR0 = *(const uint4*)gW;
    uint4 wR1 = *(const uint4*)(gW + 8);
    uint4 wR2 = *(const uint4*)(gW + 16);
    uint4 wR3 = *(const uint4*)(gW + 24);
    for (int kt = 0; kt < nkt; ++kt) {
        const int cur = kt & 1;
        *(uint4*)&As[cur][laOff]      = aR0;
        *(uint4*)&As[cur][laOff + 8]  = aR1;
        *(uint4*)&Ws[cur][lwOff]      = wR0;
        *(uint4*)&Ws[cur][lwOff + 8]  = wR1;
        *(uint4*)&Ws[cur][lwOff + 16] = wR2;
        *(uint4*)&Ws[cur][lwOff + 24] = wR3;
        __syncthreads();
        if (kt + 1 < nkt) {
            const bf16* pA = gA + (size_t)(kt + 1) * BK;
            const bf16* pW = gW + (size_t)(kt + 1) * BK;
            aR0 = *(const uint4*)pA;        aR1 = *(const uint4*)(pA + 8);
            wR0 = *(const uint4*)pW;        wR1 = *(const uint4*)(pW + 8);
            wR2 = *(const uint4*)(pW + 16); wR3 = *(const uint4*)(pW + 24);
        }
        v16bf aF[4], wF[4];
#pragma unroll
        for (int mi = 0; mi < 4; ++mi) {
            const bf16* p = &As[cur][(wm0 + mi * 16 + lh) * LSTR + hh * 8];
            v8bf lo = *(const v8bf*)p;
            v8bf hi = *(const v8bf*)(p + 16);
            aF[mi] = __builtin_shufflevector(lo, hi, 0,1,2,3,4,5,6,7,8,9,10,11,12,13,14,15);
        }
#pragma unroll
        for (int ni = 0; ni < 4; ++ni) {
            const bf16* p = &Ws[cur][(wn0 + ni * 16 + lh) * LSTR + hh * 8];
            v8bf lo = *(const v8bf*)p;
            v8bf hi = *(const v8bf*)(p + 16);
            wF[ni] = __builtin_shufflevector(lo, hi, 0,1,2,3,4,5,6,7,8,9,10,11,12,13,14,15);
        }
#pragma unroll
        for (int mi = 0; mi < 4; ++mi)
#pragma unroll
            for (int ni = 0; ni < 4; ++ni)
                acc[mi][ni] = __builtin_amdgcn_wmma_f32_16x16x32_bf16(
                    false, aF[mi], false, wF[ni], (short)0, acc[mi][ni], false, false);
        __syncthreads();
    }
#endif

    // Epilogue: C/D layout -> M = vgpr + 8*(lane>=16), N = lane&15
#pragma unroll
    for (int ni = 0; ni < 4; ++ni) {
        const int n = bn0 + wn0 + ni * 16 + lh;
        const float bz = bias[n];
#pragma unroll
        for (int mi = 0; mi < 4; ++mi) {
            const int mB = bm0 + wm0 + mi * 16 + hh * 8;
#pragma unroll
            for (int r = 0; r < 8; ++r) {
                float v = acc[mi][ni][r] + bz;
                if (RELU) v = fmaxf(v, 0.f);
                const size_t idx = (size_t)b * M * N + (size_t)(mB + r) * N + n;
                if (OUTBF) ((bf16*)outp)[idx] = (bf16)v;
                else       ((float*)outp)[idx] = v;
            }
        }
    }
}

// ---------------------------------------------------------------------------
extern "C" void kernel_launch(void* const* d_in, const int* in_sizes, int n_in,
                              void* d_out, int out_size, void* d_ws, size_t ws_size,
                              hipStream_t stream)
{
    const float* x    = (const float*)d_in[0];
    const float* mask = (const float*)d_in[1];
    const float* rw   = (const float*)d_in[2];
    const float* rb   = (const float*)d_in[3];
    const float* W1   = (const float*)d_in[4];
    const float* b1   = (const float*)d_in[5];
    const float* W2   = (const float*)d_in[6];
    const float* b2   = (const float*)d_in[7];
    float* out = (float*)d_out;

    char* ws = (char*)d_ws;
    float* part_up = (float*)(ws + 0);       // B*8*E = 512 f32
    float* part_ms = (float*)(ws + 2048);    // B*8   = 64 f32
    float* up      = (float*)(ws + 2304);    // B*E   = 64 f32
    float* mb1     = (float*)(ws + 2560);    // B*H f32
    float* mb2     = (float*)(ws + 2560 + (size_t)Bd * Hd * 4);
    size_t off = 2560 + (size_t)Bd * Hd * 4 + (size_t)Bd * Dd * 4;
    off = (off + 255) & ~(size_t)255;
    bf16* xbf   = (bf16*)(ws + off);  off += (size_t)Bd * Ld * Dd * 2;   // 32 MB
    bf16* mW1bf = (bf16*)(ws + off);  off += (size_t)Bd * Hd * Dd * 2;   // 64 MB
    bf16* mW2bf = (bf16*)(ws + off);  off += (size_t)Bd * Dd * Hd * 2;   // 64 MB
    bf16* hidbf = (bf16*)(ws + off);                                     // 128 MB

    router_kernel<<<dim3(Bd, Ld / 256), 256, 0, stream>>>(x, mask, rw, rb, part_up, part_ms);
    finalize_up_kernel<<<1, 64, 0, stream>>>(part_up, part_ms, up);
    merge_bias_kernel<<<(Bd * Hd) / 256, 256, 0, stream>>>(up, b1, mb1, Hd);
    merge_bias_kernel<<<(Bd * Dd) / 256, 256, 0, stream>>>(up, b2, mb2, Dd);
    convert_f32_bf16_kernel<<<((size_t)Bd * Ld * Dd / 4) / 256, 256, 0, stream>>>(x, xbf);
    merge_w_kernel<<<(Hd * Dd / 4) / 256, 256, 0, stream>>>(up, W1, mW1bf, Hd * Dd);
    merge_w_kernel<<<(Dd * Hd / 4) / 256, 256, 0, stream>>>(up, W2, mW2bf, Dd * Hd);

    // GEMM1: hidden = relu(xbf @ mW1^T + mb1)  -> bf16  [B][L][H]
    gemm_bf16_kernel<true, true><<<dim3(Hd / BN, Ld / BM, Bd), 256, 0, stream>>>(
        xbf, mW1bf, mb1, (void*)hidbf, Ld, Hd, Dd);
    // GEMM2: out = hidden @ mW2^T + mb2        -> f32   [B][L][D]
    gemm_bf16_kernel<false, false><<<dim3(Dd / BN, Ld / BM, Bd), 256, 0, stream>>>(
        hidbf, mW2bf, mb2, (void*)out, Ld, Dd, Hd);
}
